// LinearChainCRF_79431125172696
// MI455X (gfx1250) — matrix operand order repaired and checked by reading, not compiled, not run on hardware
//
#include <hip/hip_runtime.h>
#include <hip/hip_bf16.h>

typedef __attribute__((ext_vector_type(16))) _Float16 v16h;
typedef __attribute__((ext_vector_type(8)))  _Float16 v8h;
typedef __attribute__((ext_vector_type(8)))  float    v8f;

#define BB 128
#define TT 1024
#define CC 128

// ---------------------------------------------------------------------------
// Prep: c[j] = max_i T[i][j];  Mt[j][i] = exp(T[i][j] - c[j])  (f16, j-major
// so the WMMA B-operand lanes read 16 contiguous halves per K-chunk)
// ---------------------------------------------------------------------------
__global__ void crf_prep(const float* __restrict__ trans,
                         float* __restrict__ cmax,
                         _Float16* __restrict__ Mt) {
    int j = threadIdx.x;                 // 0..127
    float m = -INFINITY;
    for (int i = 0; i < CC; ++i) m = fmaxf(m, trans[i * CC + j]);
    cmax[j] = m;
    for (int i = 0; i < CC; ++i)
        Mt[j * CC + i] = (_Float16)__expf(trans[i * CC + j] - m);
}

// ---------------------------------------------------------------------------
// Gold path score (one block per batch element)
// ---------------------------------------------------------------------------
__global__ void crf_gold(const float* __restrict__ em,
                         const long long* __restrict__ tags,
                         const unsigned char* __restrict__ mask,
                         const float* __restrict__ trans,
                         const float* __restrict__ startT,
                         const float* __restrict__ endT,
                         float* __restrict__ gold) {
    int b = blockIdx.x;
    int tid = threadIdx.x;
    __shared__ float sred[256];
    __shared__ int   scnt[256];
    const long long* tb = tags + (long long)b * TT;
    const float*     eb = em + (size_t)b * TT * CC;
    float local = 0.f;
    int   cnt   = 0;
    for (int t = tid; t < TT; t += 256) {
        int tg = (int)tb[t];
        int mk = mask[(size_t)b * TT + t] ? 1 : 0;
        cnt += mk;
        if (t == 0) {
            local += startT[tg] + eb[tg];              // unmasked first step
        } else {
            int tp = (int)tb[t - 1];
            float v = eb[(size_t)t * CC + tg] + trans[tp * CC + tg];
            local += mk ? v : 0.f;
        }
    }
    sred[tid] = local; scnt[tid] = cnt;
    __syncthreads();
    for (int s = 128; s > 0; s >>= 1) {
        if (tid < s) { sred[tid] += sred[tid + s]; scnt[tid] += scnt[tid + s]; }
        __syncthreads();
    }
    if (tid == 0) {
        int last = scnt[0] - 1;
        gold[b] = sred[0] + endT[(int)tb[last]];
    }
}

// ---------------------------------------------------------------------------
// Forward scan (log-partition) via WMMA.
// One workgroup = 16 batch rows (M-tile), 8 waves, wave w owns j in [16w,16w+16).
// Per step: P = exp(alpha - rowmax) (f16, LDS), S = P x Mt-tile (4x wmma f16->f32),
// alpha[j] = emit + rowmax + c_j + log(S_j), masked.
// ---------------------------------------------------------------------------
__global__ __launch_bounds__(256) void crf_scan(
        const float* __restrict__ em,
        const unsigned char* __restrict__ mask,
        const float* __restrict__ startT,
        const float* __restrict__ endT,
        const float* __restrict__ cmax,
        const _Float16* __restrict__ Mt,
        float* __restrict__ logz) {
    __shared__ float              sAlpha[16 * CC];
    __shared__ __align__(32) _Float16 sP[16 * CC];
    __shared__ float              sMax[16];
    __shared__ float              sMaskF[16];

    const int tid    = threadIdx.x;
    const int w      = tid >> 5;          // wave 0..7  -> j-tile
    const int lane   = tid & 31;
    const int laneHi = lane >> 4;         // 0/1
    const int nj     = (w << 4) + (lane & 15);   // this lane's output column j
    const int b0     = blockIdx.x << 4;   // batch base

    // init alpha[bb][j] = start[j] + em[b,0,j]
    for (int idx = tid; idx < 16 * CC; idx += 256) {
        int bb = idx >> 7, j = idx & 127;
        sAlpha[idx] = startT[j] + em[((size_t)(b0 + bb) * TT) * CC + j];
    }

    // Preload B operand (Mt tile for this wave) into registers: 4 K-chunks.
    // B 32x16 layout: lane n = lane&15; lanes>=16 carry K+16; element e -> K = base+e.
    v16h bmat[4];
    {
        const _Float16* base = Mt + (size_t)nj * CC + laneHi * 16;
        #pragma unroll
        for (int kc = 0; kc < 4; ++kc) {
            const v8h* p = (const v8h*)(base + kc * 32);
            v8h lo = p[0], hi = p[1];
            #pragma unroll
            for (int q = 0; q < 8; ++q) { bmat[kc][q] = lo[q]; bmat[kc][8 + q] = hi[q]; }
        }
    }
    const float cn = cmax[nj];

    __syncthreads();

    const int arow  = tid >> 4;   // 0..15: batch row for reduction stages
    const int apart = tid & 15;   // 16 threads per row

    for (int t = 1; t < TT; ++t) {
        // Issue this step's emission loads early (independent of alpha).
        float emt[8];
        #pragma unroll
        for (int r = 0; r < 8; ++r) {
            int m = r + (laneHi << 3);
            emt[r] = em[((size_t)(b0 + m) * TT + t) * CC + nj];
        }
        // Stream-ahead prefetch (global_prefetch_b8).
        if (t + 8 < TT) {
            #pragma unroll
            for (int r = 0; r < 8; ++r) {
                int m = r + (laneHi << 3);
                __builtin_prefetch(&em[((size_t)(b0 + m) * TT + t + 8) * CC + nj], 0, 1);
            }
        }
        if (tid < 16)
            sMaskF[tid] = mask[(size_t)(b0 + tid) * TT + t] ? 1.0f : 0.0f;

        // Row max of alpha (16 threads/row, shfl_xor width-16 tree).
        float lm = -INFINITY;
        #pragma unroll
        for (int q = 0; q < 8; ++q) lm = fmaxf(lm, sAlpha[arow * CC + apart * 8 + q]);
        #pragma unroll
        for (int off = 1; off < 16; off <<= 1) lm = fmaxf(lm, __shfl_xor(lm, off, 16));
        if (apart == 0) sMax[arow] = lm;

        // P = exp(alpha - rowmax) -> f16 LDS
        #pragma unroll
        for (int q = 0; q < 8; ++q) {
            int j = apart * 8 + q;
            sP[arow * CC + j] = (_Float16)__expf(sAlpha[arow * CC + j] - lm);
        }
        __syncthreads();

        // WMMA: acc(16x16) = P(16x128) x Mt-tile(128x16), K in 4 chunks of 32.
        // A 16x32 layout: row m = lane&15; lanes>=16 carry K+16; element e -> K = base+e.
        v8f acc = {};
        const int klo = laneHi * 16;
        #pragma unroll
        for (int kc = 0; kc < 4; ++kc) {
            const _Float16* pr = sP + (lane & 15) * CC + kc * 32;
            v8h lo = *(const v8h*)(pr + klo);
            v8h hi = *(const v8h*)(pr + klo + 8);
            v16h a;
            #pragma unroll
            for (int q = 0; q < 8; ++q) { a[q] = lo[q]; a[8 + q] = hi[q]; }
            acc = __builtin_amdgcn_wmma_f32_16x16x32_f16(
                false, a, false, bmat[kc], (short)0, acc, false, false);
        }

        // alpha update. C/D layout: VGPR r -> M = r + 8*laneHi, N = lane&15.
        #pragma unroll
        for (int r = 0; r < 8; ++r) {
            int m = r + (laneHi << 3);
            float s  = fmaxf(acc[r], 1e-37f);
            float nv = emt[r] + sMax[m] + cn + __logf(s);
            float mf = sMaskF[m];
            float od = sAlpha[m * CC + nj];
            sAlpha[m * CC + nj] = mf * nv + (1.0f - mf) * od;
        }
        __syncthreads();
    }

    // logz[b] = logsumexp_j(alpha[b][j] + end[j])
    {
        float vals[8];
        float lm = -INFINITY;
        #pragma unroll
        for (int q = 0; q < 8; ++q) {
            int j = apart * 8 + q;
            vals[q] = sAlpha[arow * CC + j] + endT[j];
            lm = fmaxf(lm, vals[q]);
        }
        #pragma unroll
        for (int off = 1; off < 16; off <<= 1) lm = fmaxf(lm, __shfl_xor(lm, off, 16));
        float se = 0.f;
        #pragma unroll
        for (int q = 0; q < 8; ++q) se += __expf(vals[q] - lm);
        #pragma unroll
        for (int off = 1; off < 16; off <<= 1) se += __shfl_xor(se, off, 16);
        if (apart == 0) logz[b0 + arow] = lm + __logf(se);
    }
}

// ---------------------------------------------------------------------------
// mean(logz - gold)
// ---------------------------------------------------------------------------
__global__ void crf_final(const float* __restrict__ logz,
                          const float* __restrict__ gold,
                          float* __restrict__ out) {
    __shared__ float sred[128];
    int tid = threadIdx.x;
    sred[tid] = logz[tid] - gold[tid];
    __syncthreads();
    for (int s = 64; s > 0; s >>= 1) {
        if (tid < s) sred[tid] += sred[tid + s];
        __syncthreads();
    }
    if (tid == 0) out[0] = sred[0] / (float)BB;
}

extern "C" void kernel_launch(void* const* d_in, const int* in_sizes, int n_in,
                              void* d_out, int out_size, void* d_ws, size_t ws_size,
                              hipStream_t stream) {
    const float*         em     = (const float*)d_in[0];
    const long long*     tags   = (const long long*)d_in[1];
    const unsigned char* mask   = (const unsigned char*)d_in[2];
    const float*         trans  = (const float*)d_in[3];
    const float*         startT = (const float*)d_in[4];
    const float*         endT   = (const float*)d_in[5];

    char*     ws   = (char*)d_ws;
    float*    cmax = (float*)ws;                         // 128 f32
    _Float16* Mt   = (_Float16*)(ws + 512);              // 128*128 f16 (32 KB)
    float*    logz = (float*)(ws + 512 + 32768);         // 128 f32
    float*    gold = (float*)(ws + 512 + 32768 + 512);   // 128 f32

    crf_prep <<<1,       128, 0, stream>>>(trans, cmax, Mt);
    crf_gold <<<BB,      256, 0, stream>>>(em, tags, mask, trans, startT, endT, gold);
    crf_scan <<<BB / 16, 256, 0, stream>>>(em, mask, startT, endT, cmax, Mt, logz);
    crf_final<<<1,       128, 0, stream>>>(logz, gold, (float*)d_out);
}
